// MolAC_GCN_31233002176780
// MI455X (gfx1250) — matrix operand order repaired and checked by reading, not compiled, not run on hardware
//
#include <hip/hip_runtime.h>
#include <hip/hip_bf16.h>
#include <stdint.h>

// ---------------- problem constants ----------------
#define N_NODES   25600
#define N_EDGES   51200
#define N_GRAPH   512
#define NPG       50
#define S_STEMS   2560
#define DIM       64
#define NSTEM     105
#define CONV_STEPS 12
#define S2S_STEPS  3

// ---------------- WMMA types / helpers ----------------
typedef __attribute__((ext_vector_type(16))) __bf16       v16bf;
typedef __attribute__((ext_vector_type(8)))  float        v8f;
typedef __attribute__((ext_vector_type(4)))  unsigned int v4u;

union BF16x16 { v16bf v; v4u q[2]; unsigned short us[16]; };
union BF16x8  { v4u q;   unsigned short us[8]; };

__device__ inline unsigned short f2bf(float f) {
  unsigned int u = __float_as_uint(f);
  u += 0x7FFFu + ((u >> 16) & 1u);          // round-to-nearest-even
  return (unsigned short)(u >> 16);
}
__device__ inline float bf2f(unsigned short s) {
  return __uint_as_float(((unsigned int)s) << 16);
}
__device__ inline float lrelu(float x)    { return x > 0.f ? x : 0.01f * x; }
__device__ inline float sigmoidf_(float x){ return 1.f / (1.f + __expf(-x)); }

__device__ inline v8f wmma_bf16(v16bf a, v16bf b, v8f c) {
  return __builtin_amdgcn_wmma_f32_16x16x32_bf16(false, a, false, b, (short)0, c,
                                                 false, false);
}

// A-operand K index for 16-bit A 16x32 (ISA 7.12.2): lane half hh holds
// K in {hh*8..hh*8+7, 16+hh*8..16+hh*8+7}
__device__ inline int a_kmap(int hh, int i) {
  return (i < 8) ? (hh * 8 + i) : (16 + hh * 8 + (i - 8));
}

// ---------------- trivial kernels ----------------
__global__ void zero_kernel(float* __restrict__ p, int n) {
  int t = blockIdx.x * blockDim.x + threadIdx.x;
  if (t < n) p[t] = 0.f;
}

__global__ void deg_kernel(const int* __restrict__ dst, float* __restrict__ deg) {
  int e = blockIdx.x * blockDim.x + threadIdx.x;
  if (e < N_EDGES) atomicAdd(&deg[dst[e]], 1.f);
}

__global__ void invdeg_kernel(float* __restrict__ deg) {
  int n = blockIdx.x * blockDim.x + threadIdx.x;
  if (n < N_NODES) { float d = deg[n]; deg[n] = 1.f / (d > 1.f ? d : 1.f); }
}

// out = lrelu(x @ lin0_w.T + b)  -> h (f32 row major [N,64])
__global__ void lin0_kernel(const float* __restrict__ x, const float* __restrict__ w,
                            const float* __restrict__ b, float* __restrict__ h) {
  int t = blockIdx.x * blockDim.x + threadIdx.x;   // N*64 threads
  int n = t >> 6, o = t & 63;
  const float* xr = x + (size_t)n * 24;
  const float* wr = w + (size_t)o * 24;
  float acc = b[o];
#pragma unroll
  for (int k = 0; k < 24; ++k) acc += xr[k] * wr[k];
  h[t] = lrelu(acc);
}

// eh = lrelu(edge_attr @ net1_w.T + b) -> bf16 row major [E,128]
__global__ void eh_kernel(const float* __restrict__ ea, const float* __restrict__ w,
                          const float* __restrict__ b, unsigned short* __restrict__ eh) {
  int t = blockIdx.x * blockDim.x + threadIdx.x;   // E*128 threads
  int e = t >> 7, j = t & 127;
  const float* a  = ea + (size_t)e * 4;
  const float* wr = w + (size_t)j * 4;
  float acc = b[j] + a[0] * wr[0] + a[1] * wr[1] + a[2] * wr[2] + a[3] * wr[3];
  eh[t] = f2bf(lrelu(acc));
}

// Pack a weight matrix into the wave32 WMMA B-operand layout, bf16.
// B[k][n]: lane = hh*16 + c (c = n%16), element i -> K = kt*32 + a_kmap(hh, i).
// transposed==1: B[k][n] = src[n*ld + k];  else B[k][n] = src[k*ld + n]
__global__ void __launch_bounds__(32)
pack_b_kernel(const float* __restrict__ src, unsigned short* __restrict__ dst,
              int KT, int ld, int transposed) {
  int tile = blockIdx.x, lane = threadIdx.x;
  int nt = tile / KT, kt = tile % KT;
  int hh = lane >> 4, c = lane & 15;
  int n = nt * 16 + c;
  unsigned short* d = dst + ((size_t)tile * 32 + lane) * 16;
#pragma unroll
  for (int i = 0; i < 16; ++i) {
    int k = kt * 32 + a_kmap(hh, i);
    float v = transposed ? src[(size_t)n * ld + k] : src[(size_t)k * ld + n];
    d[i] = f2bf(v);
  }
}

// ---------------- big GEMM: W_e = eh @ net2_w.T + net2_b  (bf16 WMMA) ----
// A: eh bf16 [E,128]; B: packed net2 (KT=4, NT=256); D stored in packed
// C-tile layout: wpack[((et*256+nt)*32+lane)*8 + v] (bf16), 16B/lane stores.
__global__ void __launch_bounds__(32)
we_gemm_kernel(const unsigned short* __restrict__ eh_bf,
               const unsigned short* __restrict__ bpack,
               const float* __restrict__ bias,
               unsigned short* __restrict__ wpack) {
  int et = blockIdx.x;          // 0..3199 edge tiles of 16
  int ng = blockIdx.y;          // 0..63 groups of 4 n-tiles
  int lane = threadIdx.x, c16 = lane & 15, hh = lane >> 4;

  BF16x16 a[4];
  const unsigned short* arow = eh_bf + (size_t)(et * 16 + c16) * 128;
#pragma unroll
  for (int kt = 0; kt < 4; ++kt) {
    a[kt].q[0] = *(const v4u*)(arow + kt * 32 + hh * 8);
    a[kt].q[1] = *(const v4u*)(arow + kt * 32 + 16 + hh * 8);
  }
#pragma unroll
  for (int s = 0; s < 4; ++s) {
    int nt = ng * 4 + s;
    v8f acc = {0.f, 0.f, 0.f, 0.f, 0.f, 0.f, 0.f, 0.f};
#pragma unroll
    for (int kt = 0; kt < 4; ++kt) {
      BF16x16 b;
      const unsigned short* bp = bpack + ((size_t)(nt * 4 + kt) * 32 + lane) * 16;
      b.q[0] = *(const v4u*)bp;
      b.q[1] = *(const v4u*)(bp + 8);
      acc = wmma_bf16(a[kt].v, b.v, acc);
    }
    float bb = bias[nt * 16 + c16];
    v4u w;
#pragma unroll
    for (int v = 0; v < 4; ++v)
      w[v] = (unsigned)f2bf(acc[2 * v] + bb) |
             ((unsigned)f2bf(acc[2 * v + 1] + bb) << 16);
    *(v4u*)(wpack + ((size_t)(et * 256 + nt) * 32 + lane) * 8) = w;
  }
}

// ---------------- per-edge matvec + scatter (bandwidth kernel) ----------
// One block = one 16-edge tile; 4 waves, wave q owns output cols q*16..+15.
// agg packed layout: agg[((nodeTile*4 + nt)*32 + lane')*8 + v']
__global__ void __launch_bounds__(128)
msg_kernel(const float* __restrict__ h, const unsigned short* __restrict__ wpack,
           const int* __restrict__ src, const int* __restrict__ dst,
           float* __restrict__ agg) {
  __shared__ float xs[16 * 64];
  __shared__ int srcn[16], dstn[16];
  int et = blockIdx.x, tid = threadIdx.x;
  int e0 = et * 16;
  if (tid < 16) { srcn[tid] = src[e0 + tid]; dstn[tid] = dst[e0 + tid]; }
  __syncthreads();
  for (int idx = tid; idx < 1024; idx += 128)
    xs[idx] = h[(size_t)srcn[idx >> 6] * 64 + (idx & 63)];
  __syncthreads();

  int q = tid >> 5, lane = tid & 31, c16 = lane & 15, rbase = (lane >> 4) * 8;
  float acc[8] = {0.f, 0.f, 0.f, 0.f, 0.f, 0.f, 0.f, 0.f};
  const unsigned short* wbase = wpack + (size_t)et * 256 * 32 * 8;
  for (int d = 0; d < 64; ++d) {
    BF16x8 w;
    w.q = *(const v4u*)(wbase + ((size_t)(d * 4 + q) * 32 + lane) * 8);
#pragma unroll
    for (int v = 0; v < 8; ++v)
      acc[v] += bf2f(w.us[v]) * xs[(rbase + v) * 64 + d];
  }
#pragma unroll
  for (int v = 0; v < 8; ++v) {
    int node = dstn[rbase + v];
    int rr = node & 15;
    size_t a = ((size_t)((node >> 4) * 4 + q) * 32 + (c16 + ((rr >> 3) << 4))) * 8 +
               (rr & 7);
    atomicAdd(&agg[a], acc[v]);
  }
}

// ---------------- node update: NNConv root + bias + GRU (WMMA) -----------
__global__ void __launch_bounds__(32)
node_update_kernel(float* __restrict__ h, const float* __restrict__ agg,
                   const float* __restrict__ inv_deg,
                   const unsigned short* __restrict__ root_bp,
                   const unsigned short* __restrict__ wih_bp,
                   const unsigned short* __restrict__ whh_bp,
                   const float* __restrict__ conv_b,
                   const float* __restrict__ bih, const float* __restrict__ bhh) {
  __shared__ float lds_m[16 * 65];
  int bid = blockIdx.x, lane = threadIdx.x;
  int c16 = lane & 15, hh = lane >> 4;
  int n0 = bid * 16;

  // A operand of old hidden state (row c16), bf16, 2 K-tiles
  BF16x16 a_h[2];
  const float* hrow = h + (size_t)(n0 + c16) * 64;
#pragma unroll
  for (int kt = 0; kt < 2; ++kt)
#pragma unroll
    for (int i = 0; i < 16; ++i)
      a_h[kt].us[i] = f2bf(hrow[kt * 32 + a_kmap(hh, i)]);

  // m = lrelu(agg*inv_deg + h @ root_w + conv_b) -> LDS (16x64)
#pragma unroll
  for (int nt = 0; nt < 4; ++nt) {
    v8f acc = {0.f, 0.f, 0.f, 0.f, 0.f, 0.f, 0.f, 0.f};
#pragma unroll
    for (int kt = 0; kt < 2; ++kt) {
      BF16x16 b;
      const unsigned short* bp = root_bp + ((size_t)(nt * 2 + kt) * 32 + lane) * 16;
      b.q[0] = *(const v4u*)bp;
      b.q[1] = *(const v4u*)(bp + 8);
      acc = wmma_bf16(a_h[kt].v, b.v, acc);
    }
    float cb = conv_b[nt * 16 + c16];
    const float* ag = agg + ((size_t)(bid * 4 + nt) * 32 + lane) * 8;
#pragma unroll
    for (int v = 0; v < 8; ++v) {
      int r = v + hh * 8;
      float m = ag[v] * inv_deg[n0 + r] + acc[v] + cb;
      lds_m[r * 65 + nt * 16 + c16] = lrelu(m);
    }
  }
  __syncthreads();

  // m in A layout (bf16) via LDS transpose
  BF16x16 a_m[2];
#pragma unroll
  for (int kt = 0; kt < 2; ++kt)
#pragma unroll
    for (int i = 0; i < 16; ++i)
      a_m[kt].us[i] = f2bf(lds_m[c16 * 65 + kt * 32 + a_kmap(hh, i)]);

  // GRU gates, 16 output cols per iteration (r: nt=g, z: nt=4+g, n: nt=8+g)
#pragma unroll
  for (int g = 0; g < 4; ++g) {
    v8f accr  = {0.f, 0.f, 0.f, 0.f, 0.f, 0.f, 0.f, 0.f};
    v8f accz  = accr, accgi = accr, accgh = accr;
#pragma unroll
    for (int kt = 0; kt < 2; ++kt) {
      BF16x16 b;
      const unsigned short* p;
      p = wih_bp + ((size_t)((g)     * 2 + kt) * 32 + lane) * 16;
      b.q[0] = *(const v4u*)p; b.q[1] = *(const v4u*)(p + 8);
      accr = wmma_bf16(a_m[kt].v, b.v, accr);
      p = whh_bp + ((size_t)((g)     * 2 + kt) * 32 + lane) * 16;
      b.q[0] = *(const v4u*)p; b.q[1] = *(const v4u*)(p + 8);
      accr = wmma_bf16(a_h[kt].v, b.v, accr);
      p = wih_bp + ((size_t)((4 + g) * 2 + kt) * 32 + lane) * 16;
      b.q[0] = *(const v4u*)p; b.q[1] = *(const v4u*)(p + 8);
      accz = wmma_bf16(a_m[kt].v, b.v, accz);
      p = whh_bp + ((size_t)((4 + g) * 2 + kt) * 32 + lane) * 16;
      b.q[0] = *(const v4u*)p; b.q[1] = *(const v4u*)(p + 8);
      accz = wmma_bf16(a_h[kt].v, b.v, accz);
      p = wih_bp + ((size_t)((8 + g) * 2 + kt) * 32 + lane) * 16;
      b.q[0] = *(const v4u*)p; b.q[1] = *(const v4u*)(p + 8);
      accgi = wmma_bf16(a_m[kt].v, b.v, accgi);
      p = whh_bp + ((size_t)((8 + g) * 2 + kt) * 32 + lane) * 16;
      b.q[0] = *(const v4u*)p; b.q[1] = *(const v4u*)(p + 8);
      accgh = wmma_bf16(a_h[kt].v, b.v, accgh);
    }
    int col = g * 16 + c16;
    float br  = bih[col]       + bhh[col];
    float bz  = bih[64 + col]  + bhh[64 + col];
    float bin = bih[128 + col];
    float bhn = bhh[128 + col];
#pragma unroll
    for (int v = 0; v < 8; ++v) {
      int node = n0 + v + hh * 8;
      float rg = sigmoidf_(accr[v] + br);
      float zg = sigmoidf_(accz[v] + bz);
      float nn = tanhf((accgi[v] + bin) + rg * (accgh[v] + bhn));
      size_t idx = (size_t)node * 64 + col;
      float hold = h[idx];
      h[idx] = (1.f - zg) * nn + zg * hold;
    }
  }
}

// ---------------- stem head -------------------------------------------
__global__ void __launch_bounds__(128)
stem_kernel(const float* __restrict__ h, const int* __restrict__ stems,
            const int* __restrict__ stems_b, const float* __restrict__ w1,
            const float* __restrict__ b1, const float* __restrict__ w2,
            const float* __restrict__ b2, float* __restrict__ out) {
  __shared__ float sx[64];
  __shared__ float hid[512];
  int s = blockIdx.x, tid = threadIdx.x;
  int node = stems_b[s] * NPG + stems[s];
  if (tid < 64) sx[tid] = h[(size_t)node * 64 + tid];
  __syncthreads();
  for (int j = tid; j < 512; j += 128) {
    float acc = b1[j];
    const float* wr = w1 + (size_t)j * 64;
#pragma unroll 8
    for (int k = 0; k < 64; ++k) acc += sx[k] * wr[k];
    hid[j] = lrelu(acc);
  }
  __syncthreads();
  if (tid < NSTEM) {
    float acc = b2[tid];
    const float* wr = w2 + (size_t)tid * 512;
    for (int k = 0; k < 512; ++k) acc += hid[k] * wr[k];
    out[(size_t)s * NSTEM + tid] = acc;
  }
}

// ---------------- Set2Set ---------------------------------------------
__global__ void __launch_bounds__(256)
lstm_kernel(const float* __restrict__ wih, const float* __restrict__ whh,
            const float* __restrict__ bih, const float* __restrict__ bhh,
            float* __restrict__ qstar, float* __restrict__ hs,
            float* __restrict__ cs) {
  __shared__ float g[256];
  int b = blockIdx.x, j = threadIdx.x;
  float acc = bih[j] + bhh[j];
  const float* q  = qstar + (size_t)b * 128;
  const float* wr = wih + (size_t)j * 128;
  for (int k = 0; k < 128; ++k) acc += q[k] * wr[k];
  const float* hb  = hs + (size_t)b * 64;
  const float* wr2 = whh + (size_t)j * 64;
  for (int k = 0; k < 64; ++k) acc += hb[k] * wr2[k];
  g[j] = acc;
  __syncthreads();
  if (j < 64) {
    float ig = sigmoidf_(g[j]),       fg = sigmoidf_(g[64 + j]);
    float cg = tanhf(g[128 + j]),     og = sigmoidf_(g[192 + j]);
    float c = fg * cs[(size_t)b * 64 + j] + ig * cg;
    float hn = og * tanhf(c);
    cs[(size_t)b * 64 + j] = c;
    hs[(size_t)b * 64 + j] = hn;
    qstar[(size_t)b * 128 + j] = hn;
  }
}

__global__ void __launch_bounds__(64)
attn_kernel(const float* __restrict__ h, const float* __restrict__ hs,
            float* __restrict__ qstar) {
  __shared__ float ec[NPG];
  __shared__ float s_inv;
  int b = blockIdx.x, t = threadIdx.x;
  const float* hg  = h + (size_t)b * NPG * 64;
  const float* hsb = hs + (size_t)b * 64;
  if (t < NPG) {
    float acc = 0.f;
    const float* r = hg + (size_t)t * 64;
#pragma unroll 8
    for (int k = 0; k < 64; ++k) acc += r[k] * hsb[k];
    ec[t] = acc;
  }
  __syncthreads();
  if (t == 0) {
    float m = ec[0];
    for (int i = 1; i < NPG; ++i) m = fmaxf(m, ec[i]);
    float s = 0.f;
    for (int i = 0; i < NPG; ++i) { float a = __expf(ec[i] - m); ec[i] = a; s += a; }
    s_inv = 1.f / s;
  }
  __syncthreads();
  float inv = s_inv;
  float acc = 0.f;
  for (int i = 0; i < NPG; ++i) acc += ec[i] * hg[(size_t)i * 64 + t];
  qstar[(size_t)b * 128 + 64 + t] = acc * inv;
}

__global__ void sout_kernel(const float* __restrict__ qstar,
                            const float* __restrict__ w, const float* __restrict__ b,
                            float* __restrict__ out) {
  int bb = blockIdx.x * blockDim.x + threadIdx.x;
  if (bb >= N_GRAPH) return;
  float acc = b[0];
  const float* q = qstar + (size_t)bb * 128;
#pragma unroll 8
  for (int k = 0; k < 128; ++k) acc += q[k] * w[k];
  out[bb] = acc;
}

// ---------------- launch ----------------------------------------------
extern "C" void kernel_launch(void* const* d_in, const int* in_sizes, int n_in,
                              void* d_out, int out_size, void* d_ws, size_t ws_size,
                              hipStream_t stream) {
  (void)in_sizes; (void)n_in; (void)out_size; (void)ws_size;
  const float* x         = (const float*)d_in[0];
  const float* edge_attr = (const float*)d_in[1];
  const int*   eidx      = (const int*)d_in[2];
  const int*   stems     = (const int*)d_in[4];
  const int*   stems_b   = (const int*)d_in[5];
  const float* lin0_w = (const float*)d_in[6];
  const float* lin0_b = (const float*)d_in[7];
  const float* net1_w = (const float*)d_in[8];
  const float* net1_b = (const float*)d_in[9];
  const float* net2_w = (const float*)d_in[10];
  const float* net2_b = (const float*)d_in[11];
  const float* root_w = (const float*)d_in[12];
  const float* conv_b = (const float*)d_in[13];
  const float* gru_wih = (const float*)d_in[14];
  const float* gru_whh = (const float*)d_in[15];
  const float* gru_bih = (const float*)d_in[16];
  const float* gru_bhh = (const float*)d_in[17];
  const float* lin1_w = (const float*)d_in[18];
  const float* lin1_b = (const float*)d_in[19];
  const float* lin2_w = (const float*)d_in[20];
  const float* lin2_b = (const float*)d_in[21];
  const float* lstm_wih = (const float*)d_in[22];
  const float* lstm_whh = (const float*)d_in[23];
  const float* lstm_bih = (const float*)d_in[24];
  const float* lstm_bhh = (const float*)d_in[25];
  const float* lin3_w = (const float*)d_in[26];
  const float* lin3_b = (const float*)d_in[27];
  const int* srcA = eidx;
  const int* dstA = eidx + N_EDGES;

  // workspace carve (256B aligned); total ~447 MB
  char* base = (char*)d_ws;
  size_t off = 0;
  auto carve = [&](size_t bytes) -> void* {
    void* p = base + off;
    off += (bytes + 255) & ~(size_t)255;
    return p;
  };
  unsigned short* wpack   = (unsigned short*)carve((size_t)N_EDGES * 4096 * 2);
  unsigned short* eh_bf   = (unsigned short*)carve((size_t)N_EDGES * 128 * 2);
  unsigned short* net2_bp = (unsigned short*)carve((size_t)4 * 256 * 32 * 16 * 2);
  unsigned short* root_bp = (unsigned short*)carve((size_t)2 * 4 * 32 * 16 * 2);
  unsigned short* wih_bp  = (unsigned short*)carve((size_t)2 * 12 * 32 * 16 * 2);
  unsigned short* whh_bp  = (unsigned short*)carve((size_t)2 * 12 * 32 * 16 * 2);
  float* h       = (float*)carve((size_t)N_NODES * 64 * 4);
  float* agg     = (float*)carve((size_t)N_NODES * 64 * 4);
  float* inv_deg = (float*)carve((size_t)N_NODES * 4);
  float* hs      = (float*)carve((size_t)N_GRAPH * 64 * 4);
  float* cs      = (float*)carve((size_t)N_GRAPH * 64 * 4);
  float* qstar   = (float*)carve((size_t)N_GRAPH * 128 * 4);

  // Set2Set state zero-init (hs, cs, qstar are contiguous, 256B multiples)
  zero_kernel<<<512, 256, 0, stream>>>(hs, N_GRAPH * (64 + 64 + 128));

  lin0_kernel<<<(N_NODES * 64) / 256, 256, 0, stream>>>(x, lin0_w, lin0_b, h);
  eh_kernel<<<(N_EDGES * 128) / 256, 256, 0, stream>>>(edge_attr, net1_w, net1_b, eh_bf);

  pack_b_kernel<<<4 * 256, 32, 0, stream>>>(net2_w, net2_bp, 4, 128, 1);
  pack_b_kernel<<<2 * 4,  32, 0, stream>>>(root_w, root_bp, 2, 64, 0);
  pack_b_kernel<<<2 * 12, 32, 0, stream>>>(gru_wih, wih_bp, 2, 64, 1);
  pack_b_kernel<<<2 * 12, 32, 0, stream>>>(gru_whh, whh_bp, 2, 64, 1);

  we_gemm_kernel<<<dim3(N_EDGES / 16, 64), 32, 0, stream>>>(eh_bf, net2_bp, net2_b, wpack);

  zero_kernel<<<N_NODES / 256, 256, 0, stream>>>(inv_deg, N_NODES);
  deg_kernel<<<N_EDGES / 256, 256, 0, stream>>>(dstA, inv_deg);
  invdeg_kernel<<<N_NODES / 256, 256, 0, stream>>>(inv_deg);

  for (int step = 0; step < CONV_STEPS; ++step) {
    zero_kernel<<<(N_NODES * 64) / 256, 256, 0, stream>>>(agg, N_NODES * 64);
    msg_kernel<<<N_EDGES / 16, 128, 0, stream>>>(h, wpack, srcA, dstA, agg);
    node_update_kernel<<<N_NODES / 16, 32, 0, stream>>>(
        h, agg, inv_deg, root_bp, wih_bp, whh_bp, conv_b, gru_bih, gru_bhh);
  }

  float* out = (float*)d_out;
  stem_kernel<<<S_STEMS, 128, 0, stream>>>(h, stems, stems_b, lin1_w, lin1_b,
                                           lin2_w, lin2_b, out);
  for (int it = 0; it < S2S_STEPS; ++it) {
    lstm_kernel<<<N_GRAPH, 256, 0, stream>>>(lstm_wih, lstm_whh, lstm_bih, lstm_bhh,
                                             qstar, hs, cs);
    attn_kernel<<<N_GRAPH, 64, 0, stream>>>(h, hs, qstar);
  }
  sout_kernel<<<2, 256, 0, stream>>>(qstar, lin3_w, lin3_b,
                                     out + (size_t)S_STEMS * NSTEM);
}
